// PointCloudUpsamplerNet_52776558133920
// MI455X (gfx1250) — compile-verified, matrix operand order checked by reading, not run on hardware
//
#include <hip/hip_runtime.h>

#define N_PTS 20000
#define N_EDGE 320000
#define NOUT 80000

typedef __attribute__((ext_vector_type(16))) __bf16 v16bf;
typedef __attribute__((ext_vector_type(8))) float v8f;

union BF16Frag { v16bf v; unsigned short h[16]; unsigned int u[8]; };

__device__ __forceinline__ unsigned short f2bf(float f) {
  unsigned int u = __float_as_uint(f);
  u += 0x7fffu + ((u >> 16) & 1u);            // round-to-nearest-even
  return (unsigned short)(u >> 16);
}
__device__ __forceinline__ float bf2f(unsigned short h) {
  return __uint_as_float(((unsigned int)h) << 16);
}

// Per-lane K index of the j-th bf16 pair in a 16x32 A (or 32x16 B) fragment.
// Lane half 0: K in {0..7, 16..23}; half 1: K in {8..15, 24..31} (ISA 7.12.2).
__device__ __forceinline__ int pair_k(int j, int hf) {
  return ((j < 4) ? (2 * j) : (16 + 2 * (j - 4))) + 8 * hf;
}

// Load a fragment from a row-major bf16 row (row = M for A, N-column data for B).
// The 8 dword loads form two contiguous 16B chunks -> compiler emits b128 loads.
__device__ __forceinline__ void load_frag(BF16Frag& f, const unsigned short* row,
                                          int kbase, int hf) {
#pragma unroll
  for (int j = 0; j < 8; ++j) {
    int k = kbase + pair_k(j, hf);
    f.u[j] = *(const unsigned int*)(row + k);   // even k, 4B-aligned rows
  }
}

// Fragment of (rowj - rowi) computed in f32, repacked to bf16.
__device__ __forceinline__ void load_frag_diff(BF16Frag& f, const unsigned short* rowj,
                                               const unsigned short* rowi, int kbase, int hf) {
#pragma unroll
  for (int j = 0; j < 8; ++j) {
    int k = kbase + pair_k(j, hf);
    unsigned int uj = *(const unsigned int*)(rowj + k);
    unsigned int ui = *(const unsigned int*)(rowi + k);
    float d0 = bf2f((unsigned short)(uj & 0xffffu)) - bf2f((unsigned short)(ui & 0xffffu));
    float d1 = bf2f((unsigned short)(uj >> 16)) - bf2f((unsigned short)(ui >> 16));
    f.u[j] = (unsigned int)f2bf(d0) | ((unsigned int)f2bf(d1) << 16);
  }
}

__device__ __forceinline__ v8f wmma_bf16(const BF16Frag& a, const BF16Frag& b, v8f c) {
  return __builtin_amdgcn_wmma_f32_16x16x32_bf16(false, a.v, false, b.v, (short)0, c,
                                                 false, false);
}

// ---------------------------------------------------------------- zero init
__global__ void k_zero(float* __restrict__ p, int n) {
  int i = blockIdx.x * blockDim.x + threadIdx.x;
  if (i < n) p[i] = 0.f;
}

// --------------------------------------------- stage 1: edge MLP + segment_max
// Block = 128 threads (4 waves) = 128 edges. Layer 1 scalar -> LDS bf16,
// layer 2 via WMMA (nt outer so B frags load once), fused ReLU + atomicMax.
__global__ void __launch_bounds__(128) k_edge_mlp(
    const float* __restrict__ pts, const int* __restrict__ esrc,
    const int* __restrict__ edst, const float* __restrict__ W1a,
    const float* __restrict__ b1a, const float* __restrict__ W1b,
    const float* __restrict__ b1b, float* __restrict__ x1) {
  __shared__ float W1aS[6 * 64];
  __shared__ float b1aS[64];
  __shared__ float b1bS[64];
  __shared__ unsigned short W1bT[64 * 64];   // transposed: [n][k]
  __shared__ unsigned short h1S[128 * 72];   // padded row stride 72
  __shared__ int dstS[128];

  int tid = threadIdx.x;
  for (int i = tid; i < 6 * 64; i += 128) W1aS[i] = W1a[i];
  if (tid < 64) { b1aS[tid] = b1a[tid]; b1bS[tid] = b1b[tid]; }
  for (int i = tid; i < 64 * 64; i += 128) {
    int k = i >> 6, n = i & 63;
    W1bT[n * 64 + k] = f2bf(W1b[i]);
  }
  __syncthreads();

  int e = blockIdx.x * 128 + tid;
  int s = esrc[e], d = edst[e];
  dstS[tid] = d;
  float xi0 = pts[d * 3 + 0], xi1 = pts[d * 3 + 1], xi2 = pts[d * 3 + 2];
  float in6[6] = {xi0, xi1, xi2, pts[s * 3 + 0] - xi0, pts[s * 3 + 1] - xi1,
                  pts[s * 3 + 2] - xi2};
#pragma unroll 8
  for (int c = 0; c < 64; ++c) {
    float acc = b1aS[c];
#pragma unroll
    for (int k = 0; k < 6; ++k) acc += in6[k] * W1aS[k * 64 + c];
    h1S[tid * 72 + c] = f2bf(fmaxf(acc, 0.f));
  }
  __syncthreads();

  int wave = tid >> 5, lane = tid & 31, r = lane & 15, hf = lane >> 4;
  int* xp = (int*)x1;
  BF16Frag a[4];   // [mt][kfrag] register-resident A fragments
#pragma unroll
  for (int mt = 0; mt < 2; ++mt) {
    int mbase = (wave * 2 + mt) * 16;
    load_frag(a[mt * 2 + 0], &h1S[(mbase + r) * 72], 0, hf);
    load_frag(a[mt * 2 + 1], &h1S[(mbase + r) * 72], 32, hf);
  }
#pragma unroll
  for (int nt = 0; nt < 4; ++nt) {
    BF16Frag b0, b1;
    load_frag(b0, &W1bT[(nt * 16 + r) * 64], 0, hf);
    load_frag(b1, &W1bT[(nt * 16 + r) * 64], 32, hf);
    int col = nt * 16 + r;
#pragma unroll
    for (int mt = 0; mt < 2; ++mt) {
      int mbase = (wave * 2 + mt) * 16;
      v8f acc = {};
      acc = wmma_bf16(a[mt * 2 + 0], b0, acc);
      acc = wmma_bf16(a[mt * 2 + 1], b1, acc);
#pragma unroll
      for (int i = 0; i < 8; ++i) {
        int mrow = mbase + i + 8 * hf;
        float val = fmaxf(acc[i] + b1bS[col], 0.f);   // >= 0 -> int max == float max
        unsigned int off = (unsigned int)dstS[mrow] * 64u + (unsigned int)col;
        atomicMax(xp + off, __float_as_int(val));
      }
    }
  }
}

// ------------------------------------- stage 2: row squared-norms + bf16 copy
__global__ void __launch_bounds__(256) k_rowsq(const float* __restrict__ x1,
                                               unsigned short* __restrict__ x1b,
                                               float* __restrict__ sq) {
  int wave = threadIdx.x >> 5, lane = threadIdx.x & 31;
  int row = blockIdx.x * 8 + wave;
  float v0 = x1[row * 64 + lane];
  float v1 = x1[row * 64 + 32 + lane];
  x1b[row * 64 + lane] = f2bf(v0);
  x1b[row * 64 + 32 + lane] = f2bf(v1);
  float s = v0 * v0 + v1 * v1;
#pragma unroll
  for (int off = 16; off > 0; off >>= 1) s += __shfl_xor(s, off, 32);
  if (lane == 0) sq[row] = s;
}

// ----------------------------------------------- stage 3: kNN in feature space
// Wave owns 32 query rows (one per lane for top-K). Candidate tiles of 32:
// 8 WMMAs -> d2 = sq[n] - 2*dot; wave-uniform threshold early-out amortized
// over 32 candidates; on pass, LDS transpose so each lane sees its own row,
// then register top-16 insertion with tracked worst slot.
__global__ void __launch_bounds__(256) k_knn(const unsigned short* __restrict__ x1b,
                                             const float* __restrict__ sq,
                                             int* __restrict__ nbr) {
  __shared__ float T[8][32][33];
  int wave = threadIdx.x >> 5, lane = threadIdx.x & 31;
  int r = lane & 15, hf = lane >> 4;
  int q0 = (blockIdx.x * 8 + wave) * 32;
  if (q0 >= N_PTS) return;     // no __syncthreads in this kernel

  BF16Frag aq[4];
#pragma unroll
  for (int mt = 0; mt < 2; ++mt) {
    load_frag(aq[mt * 2 + 0], x1b + (q0 + mt * 16 + r) * 64, 0, hf);
    load_frag(aq[mt * 2 + 1], x1b + (q0 + mt * 16 + r) * 64, 32, hf);
  }

  float dlist[16];
  int ilist[16];
#pragma unroll
  for (int i = 0; i < 16; ++i) { dlist[i] = 3.4e38f; ilist[i] = 0; }
  float thr = 3.4e38f;
  int wslot = 0;

  for (int cb = 0; cb < N_PTS; cb += 32) {
    BF16Frag b0, b1, b2, b3;
    load_frag(b0, x1b + (cb + r) * 64, 0, hf);
    load_frag(b1, x1b + (cb + r) * 64, 32, hf);
    load_frag(b2, x1b + (cb + 16 + r) * 64, 0, hf);
    load_frag(b3, x1b + (cb + 16 + r) * 64, 32, hf);
    float sq0 = sq[cb + r];
    float sq1 = sq[cb + 16 + r];
    if (cb + 32 < N_PTS) {
      __builtin_prefetch(x1b + (cb + 32 + r) * 64, 0, 0);
      __builtin_prefetch(x1b + (cb + 48 + r) * 64, 0, 0);
    }

    float dv[32];
#pragma unroll
    for (int ct = 0; ct < 2; ++ct) {
#pragma unroll
      for (int mt = 0; mt < 2; ++mt) {
        v8f acc = {};
        acc = wmma_bf16(aq[mt * 2 + 0], ct ? b2 : b0, acc);
        acc = wmma_bf16(aq[mt * 2 + 1], ct ? b3 : b1, acc);
        float sqn = ct ? sq1 : sq0;
#pragma unroll
        for (int i = 0; i < 8; ++i) dv[ct * 16 + mt * 8 + i] = sqn - 2.f * acc[i];
      }
    }

    // Wave-uniform early-out: min candidate distance vs max threshold.
    float lmin = dv[0];
#pragma unroll
    for (int i = 1; i < 32; ++i) lmin = fminf(lmin, dv[i]);
    float wmin = lmin, wthr = thr;
#pragma unroll
    for (int off = 16; off > 0; off >>= 1) {
      wmin = fminf(wmin, __shfl_xor(wmin, off, 32));
      wthr = fmaxf(wthr, __shfl_xor(wthr, off, 32));
    }
    if (wmin >= wthr) continue;

    // Transpose 32x32 tile through LDS so lane owns its query row.
#pragma unroll
    for (int ct = 0; ct < 2; ++ct)
#pragma unroll
      for (int mt = 0; mt < 2; ++mt)
#pragma unroll
        for (int i = 0; i < 8; ++i)
          T[wave][mt * 16 + i + 8 * hf][ct * 16 + r] = dv[ct * 16 + mt * 8 + i];
    __builtin_amdgcn_wave_barrier();
#pragma unroll
    for (int c = 0; c < 32; ++c) {
      float v = T[wave][lane][c];
      if (v < thr) {
        int idx = cb + c;
#pragma unroll
        for (int s2 = 0; s2 < 16; ++s2)
          if (s2 == wslot) { dlist[s2] = v; ilist[s2] = idx; }
        thr = dlist[0]; wslot = 0;
#pragma unroll
        for (int s2 = 1; s2 < 16; ++s2)
          if (dlist[s2] > thr) { thr = dlist[s2]; wslot = s2; }
      }
    }
    __builtin_amdgcn_wave_barrier();
  }
  int q = q0 + lane;
#pragma unroll
  for (int c = 0; c < 16; ++c) nbr[q * 16 + c] = ilist[c];
}

// ------------------------------- stage 4: pair MLP over K=16 neighbors + max
// Wave = one node; the 16 neighbors are exactly one 16-row WMMA tile.
__global__ void __launch_bounds__(256) k_pair_mlp(
    const unsigned short* __restrict__ x1b, const int* __restrict__ nbr,
    const float* __restrict__ W2a, const float* __restrict__ b2a,
    const float* __restrict__ W2b, const float* __restrict__ b2b,
    unsigned short* __restrict__ x2b) {
  __shared__ unsigned short W2aT[64 * 128];
  __shared__ unsigned short W2bT[64 * 64];
  __shared__ float b2aS[64], b2bS[64];
  __shared__ unsigned short hS[8][16 * 72];

  int tid = threadIdx.x;
  for (int i = tid; i < 128 * 64; i += 256) {
    int k = i >> 6, n = i & 63;
    W2aT[n * 128 + k] = f2bf(W2a[i]);
  }
  for (int i = tid; i < 64 * 64; i += 256) {
    int k = i >> 6, n = i & 63;
    W2bT[n * 64 + k] = f2bf(W2b[i]);
  }
  if (tid < 64) { b2aS[tid] = b2a[tid]; b2bS[tid] = b2b[tid]; }
  __syncthreads();

  int wave = tid >> 5, lane = tid & 31, r = lane & 15, hf = lane >> 4;
  int node = blockIdx.x * 8 + wave;
  int nj = nbr[node * 16 + r];
  const unsigned short* rowi = x1b + node * 64;
  const unsigned short* rowj = x1b + nj * 64;

  BF16Frag a0, a1, a2, a3;                    // pair = [xi | xj - xi], K=128
  load_frag(a0, rowi, 0, hf);
  load_frag(a1, rowi, 32, hf);
  load_frag_diff(a2, rowj, rowi, 0, hf);
  load_frag_diff(a3, rowj, rowi, 32, hf);

#pragma unroll
  for (int nt = 0; nt < 4; ++nt) {
    BF16Frag w0, w1, w2, w3;
    const unsigned short* wrow = &W2aT[(nt * 16 + r) * 128];
    load_frag(w0, wrow, 0, hf);
    load_frag(w1, wrow, 32, hf);
    load_frag(w2, wrow, 64, hf);
    load_frag(w3, wrow, 96, hf);
    v8f acc = {};
    acc = wmma_bf16(a0, w0, acc);
    acc = wmma_bf16(a1, w1, acc);
    acc = wmma_bf16(a2, w2, acc);
    acc = wmma_bf16(a3, w3, acc);
    int col = nt * 16 + r;
#pragma unroll
    for (int i = 0; i < 8; ++i) {
      float val = fmaxf(acc[i] + b2aS[col], 0.f);
      hS[wave][(i + 8 * hf) * 72 + col] = f2bf(val);
    }
  }
  __builtin_amdgcn_wave_barrier();

  BF16Frag ha0, ha1;
  load_frag(ha0, &hS[wave][r * 72], 0, hf);
  load_frag(ha1, &hS[wave][r * 72], 32, hf);
#pragma unroll
  for (int nt = 0; nt < 4; ++nt) {
    BF16Frag w0, w1;
    const unsigned short* wrow = &W2bT[(nt * 16 + r) * 64];
    load_frag(w0, wrow, 0, hf);
    load_frag(w1, wrow, 32, hf);
    v8f acc = {};
    acc = wmma_bf16(ha0, w0, acc);
    acc = wmma_bf16(ha1, w1, acc);
    int col = nt * 16 + r;
    float m = -3.4e38f;
#pragma unroll
    for (int i = 0; i < 8; ++i) m = fmaxf(m, fmaxf(acc[i] + b2bS[col], 0.f));
    m = fmaxf(m, __shfl_xor(m, 16, 32));      // combine both halves (max over K)
    if (hf == 0) x2b[node * 64 + col] = f2bf(m);
  }
}

// ------------------------ generic bf16 GEMM: out = A(Mx64) @ W(64xncols) + b
// scatterN != 0 fuses the (N,R,64)->(R*N,64) transpose of the xe stage.
__global__ void __launch_bounds__(256) k_gemm(
    const unsigned short* __restrict__ A, const float* __restrict__ W,
    const float* __restrict__ bias, unsigned short* __restrict__ outB, int M,
    int ncols, int do_relu, int scatterN) {
  __shared__ unsigned short WT[256 * 64];   // [n][k] bf16
  __shared__ float biasS[256];
  int tid = threadIdx.x;
  int tot = 64 * ncols;
  for (int i = tid; i < tot; i += 256) {
    int k = i / ncols, n = i - k * ncols;
    WT[n * 64 + k] = f2bf(W[i]);
  }
  for (int i = tid; i < ncols; i += 256) biasS[i] = bias[i];
  __syncthreads();

  int wave = tid >> 5, lane = tid & 31, r = lane & 15, hf = lane >> 4;
  int mbase = (blockIdx.x * 8 + wave) * 16;
  if (mbase >= M) return;

  BF16Frag a0, a1;
  load_frag(a0, A + (unsigned)(mbase + r) * 64u, 0, hf);
  load_frag(a1, A + (unsigned)(mbase + r) * 64u, 32, hf);

  int ntiles = ncols >> 4;
  for (int nt = 0; nt < ntiles; ++nt) {
    BF16Frag b0, b1;
    const unsigned short* wrow = &WT[(nt * 16 + r) * 64];
    load_frag(b0, wrow, 0, hf);
    load_frag(b1, wrow, 32, hf);
    v8f acc = {};
    acc = wmma_bf16(a0, b0, acc);
    acc = wmma_bf16(a1, b1, acc);
    int col = nt * 16 + r;
#pragma unroll
    for (int i = 0; i < 8; ++i) {
      int grow = mbase + i + 8 * hf;
      float val = acc[i] + biasS[col];
      if (do_relu) val = fmaxf(val, 0.f);
      if (scatterN) {
        unsigned int orow = (unsigned int)(col >> 6) * (unsigned int)scatterN + (unsigned int)grow;
        outB[orow * 64u + (unsigned int)(col & 63)] = f2bf(val);
      } else {
        outB[(unsigned int)grow * (unsigned int)ncols + (unsigned int)col] = f2bf(val);
      }
    }
  }
}

// ------------------------------------------- final 64->3 projection (tiny)
__global__ void __launch_bounds__(256) k_final(const unsigned short* __restrict__ hb,
                                               const float* __restrict__ Wr2,
                                               const float* __restrict__ br2,
                                               float* __restrict__ out) {
  int row = blockIdx.x * 256 + threadIdx.x;
  if (row >= NOUT) return;
  float a0 = br2[0], a1 = br2[1], a2 = br2[2];
  const unsigned short* hrow = hb + (unsigned)row * 64u;
#pragma unroll 16
  for (int k = 0; k < 64; ++k) {
    float f = bf2f(hrow[k]);
    a0 += f * Wr2[k * 3 + 0];
    a1 += f * Wr2[k * 3 + 1];
    a2 += f * Wr2[k * 3 + 2];
  }
  out[row * 3 + 0] = a0;
  out[row * 3 + 1] = a1;
  out[row * 3 + 2] = a2;
}

extern "C" void kernel_launch(void* const* d_in, const int* in_sizes, int n_in,
                              void* d_out, int out_size, void* d_ws, size_t ws_size,
                              hipStream_t stream) {
  (void)in_sizes; (void)n_in; (void)out_size; (void)ws_size;
  const float* pts = (const float*)d_in[0];
  const float* W1a = (const float*)d_in[1];
  const float* b1a = (const float*)d_in[2];
  const float* W1b = (const float*)d_in[3];
  const float* b1b = (const float*)d_in[4];
  const float* W2a = (const float*)d_in[5];
  const float* b2a = (const float*)d_in[6];
  const float* W2b = (const float*)d_in[7];
  const float* b2b = (const float*)d_in[8];
  const float* We  = (const float*)d_in[9];
  const float* be  = (const float*)d_in[10];
  const float* Wp  = (const float*)d_in[11];
  const float* bp  = (const float*)d_in[12];
  const float* Wr1 = (const float*)d_in[13];
  const float* br1 = (const float*)d_in[14];
  const float* Wr2 = (const float*)d_in[15];
  const float* br2 = (const float*)d_in[16];
  const int* eidx  = (const int*)d_in[17];
  const int* esrc = eidx;           // edge_index[0]
  const int* edst = eidx + N_EDGE;  // edge_index[1]

  char* ws = (char*)d_ws;
  size_t off = 0;
  auto carve = [&](size_t bytes) -> char* {
    char* p = ws + off;
    off += (bytes + 255) & ~(size_t)255;
    return p;
  };
  float* x1            = (float*)carve((size_t)N_PTS * 64 * sizeof(float));
  unsigned short* x1b  = (unsigned short*)carve((size_t)N_PTS * 64 * 2);
  float* sq            = (float*)carve((size_t)N_PTS * sizeof(float));
  int* nbr             = (int*)carve((size_t)N_PTS * 16 * sizeof(int));
  unsigned short* x2b  = (unsigned short*)carve((size_t)N_PTS * 64 * 2);
  unsigned short* xsb  = (unsigned short*)carve((size_t)NOUT * 64 * 2);
  unsigned short* ftb  = (unsigned short*)carve((size_t)NOUT * 64 * 2);
  unsigned short* hbuf = (unsigned short*)carve((size_t)NOUT * 64 * 2);

  k_zero<<<(N_PTS * 64 + 255) / 256, 256, 0, stream>>>(x1, N_PTS * 64);
  k_edge_mlp<<<N_EDGE / 128, 128, 0, stream>>>(pts, esrc, edst, W1a, b1a, W1b, b1b, x1);
  k_rowsq<<<N_PTS / 8, 256, 0, stream>>>(x1, x1b, sq);
  k_knn<<<(N_PTS / 32 + 7) / 8, 256, 0, stream>>>(x1b, sq, nbr);
  k_pair_mlp<<<N_PTS / 8, 256, 0, stream>>>(x1b, nbr, W2a, b2a, W2b, b2b, x2b);
  // xe = x2 @ We + be, written directly in (R,N,64) scatter layout
  k_gemm<<<(N_PTS + 127) / 128, 256, 0, stream>>>(x2b, We, be, xsb, N_PTS, 256, 0, N_PTS);
  // feat = xs @ Wp + bp
  k_gemm<<<(NOUT + 127) / 128, 256, 0, stream>>>(xsb, Wp, bp, ftb, NOUT, 64, 0, 0);
  // h = relu(feat @ Wr1 + br1)
  k_gemm<<<(NOUT + 127) / 128, 256, 0, stream>>>(ftb, Wr1, br1, hbuf, NOUT, 64, 1, 0);
  // out = h @ Wr2 + br2
  k_final<<<(NOUT + 255) / 256, 256, 0, stream>>>(hbuf, Wr2, br2, (float*)d_out);
}
